// RecurrentModule_86870008529490
// MI455X (gfx1250) — matrix operand order repaired and checked by reading, not compiled, not run on hardware
//
#include <hip/hip_runtime.h>
#include <hip/hip_bf16.h>

typedef __attribute__((ext_vector_type(16))) __bf16 v16bf;
typedef __attribute__((ext_vector_type(8)))  __bf16 v8bf;
typedef __attribute__((ext_vector_type(8)))  float  v8f;
typedef __attribute__((ext_vector_type(4)))  int    v4i;

#define T_DIM 512
#define B_DIM 64
#define H_DIM 1024
#define G_DIM 3072   // 3*H

#if __has_builtin(__builtin_amdgcn_global_load_async_to_lds_b128) && \
    __has_builtin(__builtin_amdgcn_s_wait_asynccnt)
#define USE_ASYNC_LDS 1
#else
#define USE_ASYNC_LDS 0
#endif

// 16-byte global -> LDS copy: async (CDNA5 GLOBAL_LOAD_ASYNC_TO_LDS_B128,
// ASYNCcnt-tracked) when available, else VGPR bounce.
__device__ __forceinline__ void copy16_g2l(const void* g, void* l) {
#if USE_ASYNC_LDS
  __builtin_amdgcn_global_load_async_to_lds_b128(
      (__attribute__((address_space(1))) v4i*)g,
      (__attribute__((address_space(3))) v4i*)l, 0, 0);
#else
  *(uint4*)l = *(const uint4*)g;
#endif
}

#if USE_ASYNC_LDS
#define WAIT_ASYNC(n) __builtin_amdgcn_s_wait_asynccnt(n)
#else
#define WAIT_ASYNC(n) ((void)0)
#endif

__device__ __forceinline__ float sigf(float x) {
  return 1.0f / (1.0f + __expf(-x));
}
__device__ __forceinline__ float tanh_fast(float x) {
  float ax = fabsf(x);
  float e  = __expf(-2.0f * ax);
  float r  = (1.0f - e) / (1.0f + e);
  return copysignf(r, x);
}

// ---------------- fp32 -> bf16 bulk convert (8 elems/thread) ----------------
__global__ void k_cvt_bf16(const float* __restrict__ in, __bf16* __restrict__ out, int n) {
  int i = (blockIdx.x * blockDim.x + threadIdx.x) * 8;
  if (i + 8 > n) return;
  float4 a = *(const float4*)(in + i);
  float4 b = *(const float4*)(in + i + 4);
  v8bf o;
  o[0] = (__bf16)a.x; o[1] = (__bf16)a.y; o[2] = (__bf16)a.z; o[3] = (__bf16)a.w;
  o[4] = (__bf16)b.x; o[5] = (__bf16)b.y; o[6] = (__bf16)b.z; o[7] = (__bf16)b.w;
  *(v8bf*)(out + i) = o;
}

// ------------- transpose + convert: W[rows][cols] -> Wt[cols][rows] ----------
__global__ void k_transpose_cvt(const float* __restrict__ W, __bf16* __restrict__ Wt,
                                int rows, int cols) {
  __shared__ __bf16 tile[32][33];
  int c0 = blockIdx.x * 32;
  int r0 = blockIdx.y * 32;
  for (int i = threadIdx.y; i < 32; i += 8)
    tile[i][threadIdx.x] = (__bf16)W[(size_t)(r0 + i) * cols + (c0 + threadIdx.x)];
  __syncthreads();
  for (int i = threadIdx.y; i < 32; i += 8)
    Wt[(size_t)(c0 + i) * rows + (r0 + threadIdx.x)] = tile[threadIdx.x][i];
}

// ---------------- zero-init hidden state (fp32 + bf16 copies) ----------------
__global__ void k_zero_h(float* __restrict__ h, __bf16* __restrict__ hb) {
  int i = blockIdx.x * blockDim.x + threadIdx.x;   // 65536 threads
  h[i]  = 0.0f;
  hb[i] = (__bf16)0.0f;
}

// -------- gi = ins_bf16[M=T*B][K=H] @ Wt_i^T + b_i  -> bf16 [M][3H] ----------
// Block tile 128M x 128N, 8 waves; wave = 16M x 128N (8 accumulators, one A frag
// per k-step). B k-slab [128 cols x 64 k] double-buffered in LDS via async copy.
__global__ void __launch_bounds__(256)
k_gemm_gi(const __bf16* __restrict__ A, const __bf16* __restrict__ Bt,
          const float* __restrict__ bias, __bf16* __restrict__ out) {
  __shared__ __bf16 bsm[2][128 * 64];   // 2 x 16 KB
  const int lane  = threadIdx.x & 31;
  const int wave  = threadIdx.x >> 5;
  const int half  = lane >> 4;
  const int l16   = lane & 15;
  const int mbase = blockIdx.y * 128 + wave * 16;
  const int nbase = blockIdx.x * 128;

  const __bf16* arow = A + (size_t)(mbase + l16) * H_DIM;
  v8f acc[8];
#pragma unroll
  for (int j = 0; j < 8; ++j) acc[j] = v8f{};

  // stage B k-slab into LDS: layout bsm[buf][col*64 + klocal]
  auto issueB = [&](int buf, int kbase) {
#pragma unroll
    for (int s = 0; s < 4; ++s) {
      int b   = threadIdx.x * 16 + s * 4096;   // byte offset within 16 KB slab
      int col = b >> 7;                        // 128 bytes per column
      int kl  = (b & 127) >> 1;
      copy16_g2l(Bt + (size_t)(nbase + col) * H_DIM + kbase + kl,
                 (char*)&bsm[buf][0] + b);
    }
  };

  auto consume = [&](const __bf16* bs, int kbase) {
#pragma unroll
    for (int sub = 0; sub < 2; ++sub) {
      int kk = kbase + sub * 32;
      v8bf a0 = *(const v8bf*)(arow + kk + half * 8);
      v8bf a1 = *(const v8bf*)(arow + kk + 16 + half * 8);
      v16bf a = __builtin_shufflevector(a0, a1, 0,1,2,3,4,5,6,7,8,9,10,11,12,13,14,15);
#pragma unroll
      for (int j = 0; j < 8; ++j) {
        const __bf16* bp = bs + (j * 16 + l16) * 64 + sub * 32 + half * 8;
        v8bf b0 = *(const v8bf*)bp;
        v8bf b1 = *(const v8bf*)(bp + 16);
        v16bf b = __builtin_shufflevector(b0, b1, 0,1,2,3,4,5,6,7,8,9,10,11,12,13,14,15);
        acc[j] = __builtin_amdgcn_wmma_f32_16x16x32_bf16(
            false, a, false, b, (short)0, acc[j], false, false);
      }
    }
  };

  issueB(0, 0);
  for (int it = 0; it < 15; ++it) {
    issueB((it + 1) & 1, (it + 1) * 64);   // prefetch next slab
    WAIT_ASYNC(4);                         // current slab's 4 async ops done
    __syncthreads();
    consume(bsm[it & 1], it * 64);
    __syncthreads();                       // reads done before buffer reuse
  }
  WAIT_ASYNC(0);
  __syncthreads();
  consume(bsm[1], 15 * 64);

#pragma unroll
  for (int j = 0; j < 8; ++j) {
    int col = nbase + j * 16 + l16;
    float bv = bias[col];
#pragma unroll
    for (int v = 0; v < 8; ++v) {
      int row = mbase + v + half * 8;
      out[(size_t)row * G_DIM + col] = (__bf16)(acc[j][v] + bv);
    }
  }
}

// ------------------------------- GRU step ------------------------------------
// grid = 64 blocks (16-col j-tiles), block = 128 threads (4 waves = 4 M-tiles).
// Each wave: 3 gate accumulators (cols j, H+j, 2H+j of gh = h_eff @ W_h).
// Also prefetches next timestep's gi slice into L2 (off the critical path).
__global__ void __launch_bounds__(128)
k_gru_step(const float* __restrict__ h_in, const __bf16* __restrict__ hb_in,
           const __bf16* __restrict__ Wt_h, const __bf16* __restrict__ gi_t,
           const __bf16* __restrict__ gi_next,
           const int* __restrict__ reset_t, const float* __restrict__ b_hn,
           float* __restrict__ h_out, __bf16* __restrict__ hb_out,
           float* __restrict__ y_out) {
  const int lane  = threadIdx.x & 31;
  const int wave  = threadIdx.x >> 5;
  const int half  = lane >> 4;
  const int l16   = lane & 15;
  const int arow  = wave * 16 + l16;          // batch row this lane feeds into A
  const int jbase = blockIdx.x * 16;
  const bool rst  = (reset_t[arow] != 0);

  // warm L2 with next step's gi slice: 64 blk * 128 thr covers 6144 cachelines
  {
    int idx = blockIdx.x * 128 + threadIdx.x;
    if (idx < (B_DIM * G_DIM * 2) / 64)
      __builtin_prefetch((const char*)gi_next + (size_t)idx * 64, 0, 1);
  }

  const __bf16* ap = hb_in + (size_t)arow * H_DIM;
  v8f acc[3] = {v8f{}, v8f{}, v8f{}};

  for (int kk = 0; kk < H_DIM; kk += 32) {
    v8bf a0 = *(const v8bf*)(ap + kk + half * 8);
    v8bf a1 = *(const v8bf*)(ap + kk + 16 + half * 8);
    if (rst) {
#pragma unroll
      for (int i = 0; i < 8; ++i) { a0[i] = (__bf16)0.0f; a1[i] = (__bf16)0.0f; }
    }
    v16bf a = __builtin_shufflevector(a0, a1, 0,1,2,3,4,5,6,7,8,9,10,11,12,13,14,15);
#pragma unroll
    for (int g = 0; g < 3; ++g) {
      const __bf16* bp = Wt_h + (size_t)(g * H_DIM + jbase + l16) * H_DIM + kk + half * 8;
      v8bf b0 = *(const v8bf*)bp;
      v8bf b1 = *(const v8bf*)(bp + 16);
      v16bf b = __builtin_shufflevector(b0, b1, 0,1,2,3,4,5,6,7,8,9,10,11,12,13,14,15);
      acc[g] = __builtin_amdgcn_wmma_f32_16x16x32_bf16(
          false, a, false, b, (short)0, acc[g], false, false);
    }
  }

  const int col = jbase + l16;
  const float bhn = b_hn[col];
#pragma unroll
  for (int v = 0; v < 8; ++v) {
    int row = wave * 16 + v + half * 8;
    const __bf16* gir = gi_t + (size_t)row * G_DIM + col;
    float ir  = (float)gir[0];
    float iz  = (float)gir[H_DIM];
    float inn = (float)gir[2 * H_DIM];
    float hp  = (reset_t[row] != 0) ? 0.0f : h_in[(size_t)row * H_DIM + col];
    float r = sigf(ir + acc[0][v]);
    float z = sigf(iz + acc[1][v]);
    float n = tanh_fast(inn + r * (acc[2][v] + bhn));
    float nh = (1.0f - z) * n + z * hp;
    size_t o = (size_t)row * H_DIM + col;
    h_out[o]  = nh;
    hb_out[o] = (__bf16)nh;
    y_out[o]  = nh;
  }
}

// -----------------------------------------------------------------------------
extern "C" void kernel_launch(void* const* d_in, const int* in_sizes, int n_in,
                              void* d_out, int out_size, void* d_ws, size_t ws_size,
                              hipStream_t stream) {
  const float* ins    = (const float*)d_in[0];
  const int*   resets = (const int*)d_in[1];
  const float* W_i    = (const float*)d_in[2];
  const float* b_i    = (const float*)d_in[3];
  const float* W_h    = (const float*)d_in[4];
  const float* b_hn   = (const float*)d_in[5];
  float* out = (float*)d_out;

  char* ws = (char*)d_ws;
  size_t o = 0;
  __bf16* wt_i = (__bf16*)(ws + o); o += (size_t)G_DIM * H_DIM * 2;          // 6 MB
  __bf16* wt_h = (__bf16*)(ws + o); o += (size_t)G_DIM * H_DIM * 2;          // 6 MB
  __bf16* insb = (__bf16*)(ws + o); o += (size_t)T_DIM * B_DIM * H_DIM * 2;  // 64 MB
  __bf16* gib  = (__bf16*)(ws + o); o += (size_t)T_DIM * B_DIM * G_DIM * 2;  // 192 MB
  float*  h0   = (float*)(ws + o);  o += (size_t)B_DIM * H_DIM * 4;
  float*  h1   = (float*)(ws + o);  o += (size_t)B_DIM * H_DIM * 4;
  __bf16* hb0  = (__bf16*)(ws + o); o += (size_t)B_DIM * H_DIM * 2;
  __bf16* hb1  = (__bf16*)(ws + o); o += (size_t)B_DIM * H_DIM * 2;

  // 1) bulk convert ins -> bf16
  {
    int n = T_DIM * B_DIM * H_DIM;
    int blocks = (n / 8 + 255) / 256;
    k_cvt_bf16<<<blocks, 256, 0, stream>>>(ins, insb, n);
  }
  // 2) transpose+convert weights to [3H][H] bf16 (N-major for WMMA B frags)
  k_transpose_cvt<<<dim3(G_DIM / 32, H_DIM / 32), dim3(32, 8), 0, stream>>>(
      W_i, wt_i, H_DIM, G_DIM);
  k_transpose_cvt<<<dim3(G_DIM / 32, H_DIM / 32), dim3(32, 8), 0, stream>>>(
      W_h, wt_h, H_DIM, G_DIM);
  // 3) zero initial hidden state (every call: deterministic)
  k_zero_h<<<256, 256, 0, stream>>>(h0, hb0);
  // 4) big time-parallel GEMM: gi = ins @ W_i + b_i
  k_gemm_gi<<<dim3(G_DIM / 128, (T_DIM * B_DIM) / 128), 256, 0, stream>>>(
      insb, wt_i, b_i, gib);
  // 5) sequential recurrence, double-buffered hidden state
  for (int t = 0; t < T_DIM; ++t) {
    const float*  hin  = (t & 1) ? h1  : h0;
    float*        hout = (t & 1) ? h0  : h1;
    const __bf16* hbin = (t & 1) ? hb1 : hb0;
    __bf16*       hbout= (t & 1) ? hb0 : hb1;
    const __bf16* gi_t = gib + (size_t)t * B_DIM * G_DIM;
    int tn = (t + 1 < T_DIM) ? (t + 1) : t;
    const __bf16* gi_n = gib + (size_t)tn * B_DIM * G_DIM;
    k_gru_step<<<H_DIM / 16, 128, 0, stream>>>(
        hin, hbin, wt_h, gi_t, gi_n,
        resets + (size_t)t * B_DIM,
        b_hn, hout, hbout,
        out + (size_t)t * B_DIM * H_DIM);
  }
}
